// CRF_20796231647395
// MI455X (gfx1250) — compile-verified, matrix operand order checked by reading, not run on hardware
//
#include <hip/hip_runtime.h>
#include <hip/hip_bf16.h>

typedef _Float16 v16h __attribute__((ext_vector_type(16)));
typedef float    v8f  __attribute__((ext_vector_type(8)));

constexpr int Bn = 512;
constexpr int Sn = 1024;
constexpr int Tn = 64;

// ---------------------------------------------------------------------------
// Forward (partition) kernel.
// One wave handles 16 batches. State carried as normalized probabilities
// p[j] = exp(score[j] - c) in f16 WMMA-B layout, plus log-normalizer c (f32).
// Per step: y = E^T x p (WMMA, E=exp(trans) fixed in A layout),
//           q[j] = y[j] * exp(emit_t[j]*mask_t),
//           M = max_j q,  c += log M,  p = q / M.
// D-tile -> next B-operand conversion is lane-local by layout construction.
// ---------------------------------------------------------------------------
__global__ __launch_bounds__(128) void crf_forward_kernel(
    const float* __restrict__ emissions,   // (B,S,T)
    const float* __restrict__ mask,        // (B,S)
    const float* __restrict__ trans,       // (T,T)
    float* __restrict__ fwd)               // (B,)
{
  const int lane  = threadIdx.x & 31;
  const int wave  = threadIdx.x >> 5;
  const int col   = lane & 15;             // batch-in-group (N index)
  const int h     = lane >> 4;             // lane half
  const int group = blockIdx.x * 4 + wave;
  const int b     = group * 16 + col;

  // ---- Build A = exp(trans)^T, 4 M-tiles x 2 K-tiles, f16 A-layout. ----
  // A[M=j, K=i] = exp(trans[i][j]).  Element e of v16h in lane-half h:
  // K = kb*32 + 16*(e>>3) + 8*h + (e&7);  M = mb*16 + col.
  v16h A[4][2];
  #pragma unroll
  for (int mb = 0; mb < 4; ++mb) {
    #pragma unroll
    for (int kb = 0; kb < 2; ++kb) {
      v16h a;
      #pragma unroll
      for (int e = 0; e < 16; ++e) {
        const int i = kb * 32 + ((e >> 3) << 4) + h * 8 + (e & 7);
        const int j = mb * 16 + col;
        a[e] = (_Float16)__expf(trans[i * Tn + j]);
      }
      A[mb][kb] = a;
    }
  }

  const float* erow0 = emissions + ((size_t)b * Sn) * Tn;
  const float* mrow  = mask + (size_t)b * Sn;

  // ---- t = 0 : q[j] = exp(emit0[j]*mask0) ----
  // lane (col,h) owns j = 16*mb + 8*h + r  (matches f32 D-tile layout)
  float q[4][8];
  {
    const float  mk = mrow[0];
    const float4* rp = (const float4*)erow0;
    #pragma unroll
    for (int mb = 0; mb < 4; ++mb) {
      const float4 c0 = rp[4 * mb + 2 * h];
      const float4 c1 = rp[4 * mb + 2 * h + 1];
      const float ev[8] = {c0.x, c0.y, c0.z, c0.w, c1.x, c1.y, c1.z, c1.w};
      #pragma unroll
      for (int r = 0; r < 8; ++r) q[mb][r] = __expf(ev[r] * mk);
    }
  }

  float c;      // per-batch log normalizer (identical in lanes l and l^16)
  v16h  b0, b1; // p in WMMA-B layout: b0 covers K=0..31, b1 covers K=32..63
  {
    float m01 = q[0][0], m23 = q[2][0];
    #pragma unroll
    for (int r = 0; r < 8; ++r) {
      m01 = fmaxf(fmaxf(m01, q[0][r]), q[1][r]);
      m23 = fmaxf(fmaxf(m23, q[2][r]), q[3][r]);
    }
    float M = fmaxf(m01, m23);
    M = fmaxf(M, __shfl_xor(M, 16));
    c = __logf(M);
    const float rm = __builtin_amdgcn_rcpf(M);
    #pragma unroll
    for (int e = 0; e < 16; ++e) {
      b0[e] = (_Float16)(q[(e >> 3)    ][e & 7] * rm);
      b1[e] = (_Float16)(q[(e >> 3) + 2][e & 7] * rm);
    }
  }

  // ---- Sequential scan over t = 1..S-1 ----
  for (int t = 1; t < Sn; ++t) {
    const float* erow = erow0 + (size_t)t * Tn;
    // WGP(CU)-scope prefetch (default scope = lowest -> fills L0/WGP$),
    // 4 rows (1 KB / ~4 iterations) ahead; lane-half offset covers both
    // 128B cache lines of each batch's 256B row. __builtin_prefetch only
    // reaches SE scope, which per ISA 10.5 does NOT fill the WGP cache.
    {
      const float* pf = erow + 4 * Tn + h * 32;
      asm volatile("global_prefetch_b8 %0, off" :: "v"(pf));
    }

    // y = E^T x p : 2 WMMAs (K=64) per 16x16 output tile, 4 tiles.
    v8f y[4];
    #pragma unroll
    for (int mb = 0; mb < 4; ++mb) {
      v8f z = {};
      z = __builtin_amdgcn_wmma_f32_16x16x32_f16(false, A[mb][0], false, b0,
                                                 (short)0, z, false, false);
      z = __builtin_amdgcn_wmma_f32_16x16x32_f16(false, A[mb][1], false, b1,
                                                 (short)0, z, false, false);
      y[mb] = z;
    }

    const float  mk = mrow[t];
    const float4* rp = (const float4*)erow;
    #pragma unroll
    for (int mb = 0; mb < 4; ++mb) {
      const float4 c0 = rp[4 * mb + 2 * h];
      const float4 c1 = rp[4 * mb + 2 * h + 1];
      const float ev[8] = {c0.x, c0.y, c0.z, c0.w, c1.x, c1.y, c1.z, c1.w};
      #pragma unroll
      for (int r = 0; r < 8; ++r) q[mb][r] = y[mb][r] * __expf(ev[r] * mk);
    }

    float m01 = q[0][0], m23 = q[2][0];
    #pragma unroll
    for (int r = 0; r < 8; ++r) {
      m01 = fmaxf(fmaxf(m01, q[0][r]), q[1][r]);
      m23 = fmaxf(fmaxf(m23, q[2][r]), q[3][r]);
    }
    float M = fmaxf(m01, m23);
    M = fmaxf(M, __shfl_xor(M, 16));          // combine the two lane-halves
    c += __logf(M);                            // 1 log per batch per step
    const float rm = __builtin_amdgcn_rcpf(M);
    #pragma unroll
    for (int e = 0; e < 16; ++e) {
      b0[e] = (_Float16)(q[(e >> 3)    ][e & 7] * rm);
      b1[e] = (_Float16)(q[(e >> 3) + 2][e & 7] * rm);
    }
  }

  // ---- logZ = c + log(sum_j p[j]) ----
  float sp = 0.f;
  #pragma unroll
  for (int e = 0; e < 16; ++e) sp += (float)b0[e] + (float)b1[e];
  sp += __shfl_xor(sp, 16);
  if (h == 0) fwd[b] = c + __logf(sp);
}

// ---------------------------------------------------------------------------
// Gold score: one block per batch, gather + reduce.
// ---------------------------------------------------------------------------
__global__ __launch_bounds__(256) void crf_gold_kernel(
    const float* __restrict__ emissions,
    const int*   __restrict__ tags,
    const float* __restrict__ mask,
    const float* __restrict__ trans,
    float* __restrict__ gold)
{
  const int b = blockIdx.x;
  __shared__ float sm[256];
  float acc = 0.f;
  for (int t = 1 + (int)threadIdx.x; t < Sn; t += 256) {
    const int tt = tags[b * Sn + t];
    const int tp = tags[b * Sn + t - 1];
    const float e = emissions[((size_t)b * Sn + t) * Tn + tt];
    acc += (trans[tt * Tn + tp] + e) * mask[b * Sn + t];
  }
  if (threadIdx.x == 0) {
    const int t0 = tags[b * Sn];
    acc += emissions[(size_t)b * Sn * Tn + t0] * mask[b * Sn];
  }
  sm[threadIdx.x] = acc;
  __syncthreads();
  for (int s = 128; s > 0; s >>= 1) {
    if ((int)threadIdx.x < s) sm[threadIdx.x] += sm[threadIdx.x + s];
    __syncthreads();
  }
  if (threadIdx.x == 0) gold[b] = sm[0];
}

// ---------------------------------------------------------------------------
// Final: mean(fwd - gold) -> scalar.
// ---------------------------------------------------------------------------
__global__ __launch_bounds__(256) void crf_reduce_kernel(
    const float* __restrict__ fwd,
    const float* __restrict__ gold,
    float* __restrict__ out)
{
  __shared__ float sm[256];
  float a = 0.f;
  for (int i = threadIdx.x; i < Bn; i += 256) a += fwd[i] - gold[i];
  sm[threadIdx.x] = a;
  __syncthreads();
  for (int s = 128; s > 0; s >>= 1) {
    if ((int)threadIdx.x < s) sm[threadIdx.x] += sm[threadIdx.x + s];
    __syncthreads();
  }
  if (threadIdx.x == 0) out[0] = sm[0] / (float)Bn;
}

extern "C" void kernel_launch(void* const* d_in, const int* in_sizes, int n_in,
                              void* d_out, int out_size, void* d_ws, size_t ws_size,
                              hipStream_t stream) {
  const float* emissions = (const float*)d_in[0];   // (B,S,T) f32
  const int*   tags      = (const int*)  d_in[1];   // (B,S)   i32
  const float* mask      = (const float*)d_in[2];   // (B,S)   f32
  const float* trans     = (const float*)d_in[3];   // (T,T)   f32
  float* out  = (float*)d_out;
  float* fwd  = (float*)d_ws;        // Bn floats
  float* gold = fwd + Bn;            // Bn floats

  // 32 batch-groups of 16, 4 waves (groups) per 128-thread block.
  crf_forward_kernel<<<dim3(Bn / 64), dim3(128), 0, stream>>>(emissions, mask, trans, fwd);
  crf_gold_kernel  <<<dim3(Bn),      dim3(256), 0, stream>>>(emissions, tags, mask, trans, gold);
  crf_reduce_kernel<<<1,             dim3(256), 0, stream>>>(fwd, gold, out);
}
